// PointNetEstimationv2_12781822673510
// MI455X (gfx1250) — matrix operand order repaired and checked by reading, not compiled, not run on hardware
//
#include <hip/hip_runtime.h>
#include <cstddef>
#include <cstdint>

#define EPS_F     1e-5f
#define RADIUS_F  0.2f
#define R2_F      (RADIUS_F * RADIUS_F)
#define INV_R     5.0f
#define NSAMPLE   32
#define NPOINT    256
#define BATCH     16
#define NPTS      16384
#define MTOT      (BATCH * NPOINT * NSAMPLE)   // 131072 points through MLP

typedef __attribute__((ext_vector_type(8)))  _Float16 v8h;
typedef __attribute__((ext_vector_type(16))) _Float16 v16h;
typedef __attribute__((ext_vector_type(8)))  float    v8f;

// ---------------------------------------------------------------------------
// Prep: fold BN into scale/shift, convert weights to f16 with K zero-padding.
// ---------------------------------------------------------------------------
__global__ void prep_layer(const float* __restrict__ W, const float* __restrict__ bias,
                           const float* __restrict__ g, const float* __restrict__ beta,
                           const float* __restrict__ mean, const float* __restrict__ var,
                           _Float16* __restrict__ W16, float* __restrict__ scale,
                           float* __restrict__ shift, int out_d, int in_d, int kpad) {
  int i = blockIdx.x * blockDim.x + threadIdx.x;
  int total = out_d * kpad;
  if (i < total) {
    int o = i / kpad, k = i % kpad;
    W16[i] = (_Float16)(k < in_d ? W[o * in_d + k] : 0.0f);
  }
  if (i < out_d) {
    float sc = g[i] * rsqrtf(var[i] + EPS_F);
    scale[i] = sc;
    shift[i] = (bias[i] - mean[i]) * sc + beta[i];
  }
}

// ---------------------------------------------------------------------------
// FPS: one block per batch; 256 serial steps; register-resident min-distances.
// pts layout is (B,3,N) SoA. Contiguous point ownership keeps argmax ties in
// ascending-index order, matching jnp.argmax.
// ---------------------------------------------------------------------------
__global__ __launch_bounds__(1024) void fps_kernel(const float* __restrict__ pts,
                                                   int* __restrict__ fpsIdx,
                                                   float* __restrict__ newXyz) {
  const int b = blockIdx.x, tid = threadIdx.x;
  const int lane = tid & 31, wid = tid >> 5;
  const float* px = pts + (size_t)b * 3 * NPTS;
  const float* py = px + NPTS;
  const float* pz = px + 2 * NPTS;
  __shared__ float scx, scy, scz;
  __shared__ int   slast;
  __shared__ float wbest[32];
  __shared__ int   warg[32];
  if (tid == 0) { slast = 0; scx = px[0]; scy = py[0]; scz = pz[0]; }
  float dist[16];
#pragma unroll
  for (int i = 0; i < 16; ++i) dist[i] = 1e10f;
  const int n0 = tid * 16;
  __syncthreads();
  for (int s = 0; s < NPOINT; ++s) {
    const float cx = scx, cy = scy, cz = scz;
    if (tid == 0) {
      int g = b * NPOINT + s;
      fpsIdx[g] = slast;
      newXyz[g * 3 + 0] = cx; newXyz[g * 3 + 1] = cy; newXyz[g * 3 + 2] = cz;
    }
    float best = -1.0f; int bi = n0;
#pragma unroll
    for (int i = 0; i < 16; ++i) {
      int n = n0 + i;
      float dx = px[n] - cx, dy = py[n] - cy, dz = pz[n] - cz;
      float dd = fminf(dist[i], dx * dx + dy * dy + dz * dz);
      dist[i] = dd;
      if (dd > best) { best = dd; bi = n; }
    }
#pragma unroll
    for (int off = 16; off > 0; off >>= 1) {
      float ov = __shfl_down(best, off, 32);
      int   oi = __shfl_down(bi,   off, 32);
      if (ov > best) { best = ov; bi = oi; }
    }
    if (lane == 0) { wbest[wid] = best; warg[wid] = bi; }
    __syncthreads();
    if (wid == 0) {
      best = wbest[lane]; bi = warg[lane];
#pragma unroll
      for (int off = 16; off > 0; off >>= 1) {
        float ov = __shfl_down(best, off, 32);
        int   oi = __shfl_down(bi,   off, 32);
        if (ov > best) { best = ov; bi = oi; }
      }
      if (lane == 0) { slast = bi; scx = px[bi]; scy = py[bi]; scz = pz[bi]; }
    }
    __syncthreads();
  }
}

// ---------------------------------------------------------------------------
// Ball query: one wave per center; ballot + prefix-popcount appends the 32
// lowest-index in-ball points (reference top_k(-index) semantics), fill with
// the first in-ball index.
// ---------------------------------------------------------------------------
__global__ __launch_bounds__(128) void ball_query_kernel(const float* __restrict__ pts,
                                                         const float* __restrict__ newXyz,
                                                         int* __restrict__ ballIdx) {
  const int wid  = blockIdx.x * (blockDim.x >> 5) + (threadIdx.x >> 5);
  const int lane = threadIdx.x & 31;
  const int b = wid >> 8;  // 256 centers per batch
  const float* c = newXyz + (size_t)wid * 3;
  const float cx = c[0], cy = c[1], cz = c[2];
  const float* px = pts + (size_t)b * 3 * NPTS;
  const float* py = px + NPTS;
  const float* pz = px + 2 * NPTS;
  int* out = ballIdx + (size_t)wid * NSAMPLE;
  int cnt = 0, firstIdx = -1;
  for (int base = 0; base < NPTS && cnt < NSAMPLE; base += 32) {
    int n = base + lane;
    float dx = px[n] - cx, dy = py[n] - cy, dz = pz[n] - cz;
    bool pred = (dx * dx + dy * dy + dz * dz) < R2_F;
    unsigned mask = (unsigned)__ballot(pred);
    if (firstIdx < 0 && mask) firstIdx = base + __ffs(mask) - 1;
    int pre = __popc(mask & ((1u << lane) - 1u));
    if (pred && (cnt + pre) < NSAMPLE) out[cnt + pre] = n;
    cnt += __popc(mask);
  }
  if (cnt < NSAMPLE) {
    if (firstIdx < 0) firstIdx = 0;
    for (int j = cnt + lane; j < NSAMPLE; j += 32) out[j] = firstIdx;
  }
}

// ---------------------------------------------------------------------------
// Gather + normalize -> X0 f16 [MTOT x 32] (3 valid channels, zero-padded K).
// ---------------------------------------------------------------------------
__global__ void group_feat_kernel(const float* __restrict__ pts,
                                  const int* __restrict__ ballIdx,
                                  const float* __restrict__ newXyz,
                                  _Float16* __restrict__ X0) {
  int i = blockIdx.x * blockDim.x + threadIdx.x;
  if (i >= MTOT) return;
  int b = i >> 13;             // 8192 points per batch
  int g = i >> 5;              // global group id b*256+s
  int n = ballIdx[i];
  const float* px = pts + (size_t)b * 3 * NPTS;
  const float* c  = newXyz + (size_t)g * 3;
  v8h f = {};
  f[0] = (_Float16)((px[n]            - c[0]) * INV_R);
  f[1] = (_Float16)((px[NPTS + n]     - c[1]) * INV_R);
  f[2] = (_Float16)((px[2 * NPTS + n] - c[2]) * INV_R);
  v8h z = {};
  _Float16* row = X0 + (size_t)i * 32;
  *(v8h*)(row)      = f;
  *(v8h*)(row + 8)  = z;
  *(v8h*)(row + 16) = z;
  *(v8h*)(row + 24) = z;
}

// ---------------------------------------------------------------------------
// WMMA GEMM + folded BN + ReLU.  Y[m][o] = relu(scale[o]*sum_k X[m][k]W16[o][k]
// + shift[o]).  One 16x16 output tile per wave, K-loop of v_wmma_f32_16x16x32_f16.
// fuseMax: atomic uint-max (exact for non-negative floats) into group features.
// ---------------------------------------------------------------------------
__global__ __launch_bounds__(128) void gemm_bn_relu(const _Float16* __restrict__ X,
                                                    const _Float16* __restrict__ W16,
                                                    const float* __restrict__ scale,
                                                    const float* __restrict__ shift,
                                                    _Float16* __restrict__ Y,
                                                    float* __restrict__ gmax,
                                                    int K, int Cout, int tilesN, int fuseMax) {
  const int wave = blockIdx.x * (blockDim.x >> 5) + (threadIdx.x >> 5);
  const int lane = threadIdx.x & 31;
  const int tilesM = MTOT / 16;
  const int tm = wave / tilesN, tn = wave % tilesN;
  if (tm >= tilesM) return;
  const int half = lane >> 4, mr = lane & 15;
  const _Float16* xrow = X   + (size_t)(tm * 16 + mr) * K;   // A: lane = row M
  const _Float16* wrow = W16 + (size_t)(tn * 16 + mr) * K;   // B: lane = column N
  v8f acc = {};
  for (int k0 = 0; k0 < K; k0 += 32) {
    union { v16h v; v8h h[2]; } a;
    a.h[0] = *(const v8h*)(xrow + k0 + half * 8);        // K: 0-7 / 8-15
    a.h[1] = *(const v8h*)(xrow + k0 + 16 + half * 8);   // K: 16-23 / 24-31
    union { v16h v; v8h h[2]; } bu;
    bu.h[0] = *(const v8h*)(wrow + k0 + half * 8);       // K: 0-7 / 8-15
    bu.h[1] = *(const v8h*)(wrow + k0 + 16 + half * 8);  // K: 16-23 / 24-31
    v16h bfrag = bu.v;
    acc = __builtin_amdgcn_wmma_f32_16x16x32_f16(false, a.v, false, bfrag,
                                                 (short)0, acc, false, false);
  }
  const int c = tn * 16 + mr;
  const float sc = scale[c], sh = shift[c];
#pragma unroll
  for (int r = 0; r < 8; ++r) {
    int m = tm * 16 + r + half * 8;   // C/D: VGPR r -> row r (+8 for hi lanes)
    float y = acc[r] * sc + sh;
    y = y > 0.0f ? y : 0.0f;
    if (fuseMax) {
      unsigned* dst = (unsigned*)&gmax[(size_t)(m >> 5) * Cout + c];
      atomicMax(dst, __float_as_uint(y));   // exact max: all values >= 0
    } else {
      Y[(size_t)m * Cout + c] = (_Float16)y;
    }
  }
}

__global__ void zero_f32(float* __restrict__ p, int n) {
  int i = blockIdx.x * blockDim.x + threadIdx.x;
  if (i < n) p[i] = 0.0f;
}

// max over 256 groups -> global feature (16 x 512)
__global__ void reduce_groups(const float* __restrict__ gfeat, float* __restrict__ gout) {
  int i = blockIdx.x * blockDim.x + threadIdx.x;
  if (i >= BATCH * 512) return;
  int b = i >> 9, c = i & 511;
  const float* p = gfeat + (size_t)b * NPOINT * 512 + c;
  float m = 0.0f;
  for (int s = 0; s < NPOINT; ++s) m = fmaxf(m, p[(size_t)s * 512]);
  gout[i] = m;
}

// ---------------------------------------------------------------------------
// Tiny FC head (16 rows), scalar f32, single block. Global scratch with
// threadfence+barrier between stages.
// ---------------------------------------------------------------------------
__device__ __forceinline__ float bn1f(float x, const float* g, const float* be,
                                      const float* m, const float* v, int c) {
  return (x - m[c]) * rsqrtf(v[c] + EPS_F) * g[c] + be[c];
}

__global__ __launch_bounds__(256) void head_kernel(
    const float* __restrict__ gfeat,
    const float* __restrict__ oneHot, const float* __restrict__ center,
    const float* __restrict__ Wc, const float* __restrict__ bc,
    const float* __restrict__ Wd, const float* __restrict__ bd,
    const float* __restrict__ dg, const float* __restrict__ dbe,
    const float* __restrict__ dm, const float* __restrict__ dv,
    const float* __restrict__ W1, const float* __restrict__ b1,
    const float* __restrict__ g1, const float* __restrict__ be1,
    const float* __restrict__ m1, const float* __restrict__ v1,
    const float* __restrict__ W2, const float* __restrict__ b2,
    const float* __restrict__ g2, const float* __restrict__ be2,
    const float* __restrict__ m2, const float* __restrict__ v2,
    const float* __restrict__ W3, const float* __restrict__ b3,
    const float* __restrict__ g3, const float* __restrict__ be3,
    const float* __restrict__ m3, const float* __restrict__ v3,
    const float* __restrict__ W4, const float* __restrict__ b4,
    float* __restrict__ xcat, float* __restrict__ h1,
    float* __restrict__ h2, float* __restrict__ h3,
    float* __restrict__ out) {
  const int tid = threadIdx.x;
  // concat: [global_feat(512) | relu(oh@Wc^T+bc)(64) | relu(bn(ce@Wd^T+bd))(64)]
  for (int i = tid; i < BATCH * 512; i += 256)
    xcat[(i >> 9) * 640 + (i & 511)] = gfeat[i];
  for (int i = tid; i < BATCH * 64; i += 256) {
    int b = i >> 6, c = i & 63;
    float s = bc[c], t = bd[c];
    for (int k = 0; k < 3; ++k) {
      s += oneHot[b * 3 + k] * Wc[c * 3 + k];
      t += center[b * 3 + k] * Wd[c * 3 + k];
    }
    xcat[b * 640 + 512 + c] = fmaxf(s, 0.0f);
    t = bn1f(t, dg, dbe, dm, dv, c);
    xcat[b * 640 + 576 + c] = fmaxf(t, 0.0f);
  }
  __threadfence(); __syncthreads();
  for (int i = tid; i < BATCH * 512; i += 256) {            // fc1: 640 -> 512
    int b = i >> 9, o = i & 511;
    float a = b1[o];
    for (int k = 0; k < 640; ++k) a += xcat[b * 640 + k] * W1[o * 640 + k];
    h1[i] = fmaxf(bn1f(a, g1, be1, m1, v1, o), 0.0f);
  }
  __threadfence(); __syncthreads();
  for (int i = tid; i < BATCH * 256; i += 256) {            // fc2: 512 -> 256
    int b = i >> 8, o = i & 255;
    float a = b2[o];
    for (int k = 0; k < 512; ++k) a += h1[b * 512 + k] * W2[o * 512 + k];
    h2[i] = fmaxf(bn1f(a, g2, be2, m2, v2, o), 0.0f);
  }
  __threadfence(); __syncthreads();
  for (int i = tid; i < BATCH * 64; i += 256) {             // fc3: 256 -> 64
    int b = i >> 6, o = i & 63;
    float a = b3[o];
    for (int k = 0; k < 256; ++k) a += h2[b * 256 + k] * W3[o * 256 + k];
    h3[i] = fmaxf(bn1f(a, g3, be3, m3, v3, o), 0.0f);
  }
  __threadfence(); __syncthreads();
  for (int i = tid; i < BATCH * 7; i += 256) {              // fc4: 64 -> 7
    int b = i / 7, o = i % 7;
    float a = b4[o];
    for (int k = 0; k < 64; ++k) a += h3[b * 64 + k] * W4[o * 64 + k];
    out[i] = a;
  }
}

// ---------------------------------------------------------------------------
extern "C" void kernel_launch(void* const* d_in, const int* in_sizes, int n_in,
                              void* d_out, int out_size, void* d_ws, size_t ws_size,
                              hipStream_t stream) {
  const float* pts      = (const float*)d_in[0];
  const float* oneHot   = (const float*)d_in[1];
  const float* center   = (const float*)d_in[2];
  // mlp_params: 4 layers x (W,b,g,beta,mean,var) at indices 3..26
  const float* L[4][6];
  for (int l = 0; l < 4; ++l)
    for (int j = 0; j < 6; ++j) L[l][j] = (const float*)d_in[3 + l * 6 + j];
  const float* Wc = (const float*)d_in[27];  const float* bc = (const float*)d_in[28];
  const float* Wd = (const float*)d_in[29];  const float* bd = (const float*)d_in[30];
  const float* dg = (const float*)d_in[31];  const float* dbe = (const float*)d_in[32];
  const float* dm = (const float*)d_in[33];  const float* dv = (const float*)d_in[34];
  const float* W1 = (const float*)d_in[35];  const float* b1 = (const float*)d_in[36];
  const float* g1 = (const float*)d_in[37];  const float* be1 = (const float*)d_in[38];
  const float* m1 = (const float*)d_in[39];  const float* v1 = (const float*)d_in[40];
  const float* W2 = (const float*)d_in[41];  const float* b2 = (const float*)d_in[42];
  const float* g2 = (const float*)d_in[43];  const float* be2 = (const float*)d_in[44];
  const float* m2 = (const float*)d_in[45];  const float* v2 = (const float*)d_in[46];
  const float* W3 = (const float*)d_in[47];  const float* b3 = (const float*)d_in[48];
  const float* g3 = (const float*)d_in[49];  const float* be3 = (const float*)d_in[50];
  const float* m3 = (const float*)d_in[51];  const float* v3 = (const float*)d_in[52];
  const float* W4 = (const float*)d_in[53];  const float* b4 = (const float*)d_in[54];

  char* ws = (char*)d_ws;
  size_t off = 0;
  auto alloc = [&](size_t bytes) -> void* {
    void* p = ws + off;
    off += (bytes + 255) & ~size_t(255);
    return p;
  };
  int*      fpsIdx  = (int*)alloc(BATCH * NPOINT * 4);
  float*    newXyz  = (float*)alloc(BATCH * NPOINT * 3 * 4);
  int*      ballIdx = (int*)alloc((size_t)MTOT * 4);
  static const int dims[5]  = {3, 64, 128, 256, 512};
  static const int kpads[5] = {32, 32, 64, 128, 256};   // padded K per layer
  _Float16* W16[4]; float* scale[4]; float* shift[4];
  for (int l = 0; l < 4; ++l) {
    int out_d = dims[l + 1], kp = kpads[l + 1] == 0 ? 32 : ((l == 0) ? 32 : dims[l]);
    (void)kp;
    int kpad = (l == 0) ? 32 : dims[l];
    W16[l]   = (_Float16*)alloc((size_t)out_d * kpad * 2);
    scale[l] = (float*)alloc(out_d * 4);
    shift[l] = (float*)alloc(out_d * 4);
  }
  _Float16* X0 = (_Float16*)alloc((size_t)MTOT * 32 * 2);
  _Float16* A1 = (_Float16*)alloc((size_t)MTOT * 64 * 2);
  _Float16* A2 = (_Float16*)alloc((size_t)MTOT * 128 * 2);
  _Float16* A3 = (_Float16*)alloc((size_t)MTOT * 256 * 2);
  float* gfeat = (float*)alloc((size_t)BATCH * NPOINT * 512 * 4);
  float* gout  = (float*)alloc(BATCH * 512 * 4);
  float* xcat  = (float*)alloc(BATCH * 640 * 4);
  float* h1    = (float*)alloc(BATCH * 512 * 4);
  float* h2    = (float*)alloc(BATCH * 256 * 4);
  float* h3    = (float*)alloc(BATCH * 64 * 4);

  // 1) weight/BN prep
  for (int l = 0; l < 4; ++l) {
    int out_d = dims[l + 1], in_d = dims[l], kpad = (l == 0) ? 32 : dims[l];
    int total = out_d * kpad;
    prep_layer<<<(total + 255) / 256, 256, 0, stream>>>(
        L[l][0], L[l][1], L[l][2], L[l][3], L[l][4], L[l][5],
        W16[l], scale[l], shift[l], out_d, in_d, kpad);
  }
  // 2) FPS (serial scan, one block per batch)
  fps_kernel<<<BATCH, 1024, 0, stream>>>(pts, fpsIdx, newXyz);
  // 3) ball query (one wave per center)
  ball_query_kernel<<<(BATCH * NPOINT) / 4, 128, 0, stream>>>(pts, newXyz, ballIdx);
  // 4) gather + normalize into padded f16 activations
  group_feat_kernel<<<MTOT / 256, 256, 0, stream>>>(pts, ballIdx, newXyz, X0);
  // 5) zero group-max accumulator (0.0f == uint 0, valid for non-neg atomic max)
  {
    int n = BATCH * NPOINT * 512;
    zero_f32<<<(n + 255) / 256, 256, 0, stream>>>(gfeat, n);
  }
  // 6) WMMA MLP: 3(pad32)->64->128->256->512, layer4 fuses max-over-samples
  const int tilesM = MTOT / 16;
  gemm_bn_relu<<<tilesM * 4  / 4, 128, 0, stream>>>(X0, W16[0], scale[0], shift[0], A1, gfeat,  32,  64,  4, 0);
  gemm_bn_relu<<<tilesM * 8  / 4, 128, 0, stream>>>(A1, W16[1], scale[1], shift[1], A2, gfeat,  64, 128,  8, 0);
  gemm_bn_relu<<<tilesM * 16 / 4, 128, 0, stream>>>(A2, W16[2], scale[2], shift[2], A3, gfeat, 128, 256, 16, 0);
  gemm_bn_relu<<<tilesM * 32 / 4, 128, 0, stream>>>(A3, W16[3], scale[3], shift[3], (_Float16*)nullptr, gfeat, 256, 512, 32, 1);
  // 7) max over 256 groups -> global feature
  reduce_groups<<<(BATCH * 512 + 255) / 256, 256, 0, stream>>>(gfeat, gout);
  // 8) FC head
  head_kernel<<<1, 256, 0, stream>>>(gout, oneHot, center,
                                     Wc, bc, Wd, bd, dg, dbe, dm, dv,
                                     W1, b1, g1, be1, m1, v1,
                                     W2, b2, g2, be2, m2, v2,
                                     W3, b3, g3, be3, m3, v3,
                                     W4, b4,
                                     xcat, h1, h2, h3, (float*)d_out);
}
